// InformedOpenUnmix3_28312424415860
// MI455X (gfx1250) — matrix operand order repaired and checked by reading, not compiled, not run on hardware
//
#include <hip/hip_runtime.h>
#include <math.h>

typedef __attribute__((ext_vector_type(16))) _Float16 v16h;
typedef __attribute__((ext_vector_type(8)))  _Float16 v8h;
typedef __attribute__((ext_vector_type(8)))  float    v8f;

#define NFFT 512
#define HOPK 256
#define NB   257
#define NBP  288      // 257 padded to 18*16 (also fc1 K padded to 9*32)
#define HIDN 512
#define H2   256
#define G4   1024
#define LTXT 512
#define NVOC 44
#define NFR  2048
#define NBAT 8
#define FB   (NFR*NBAT)
#define TAUD 524544
#define NEGV -100000.0f
#define POSV  200000.0f

__device__ __forceinline__ float sigf(float x){ return 1.f/(1.f+__expf(-x)); }

// ---- WMMA fragment loaders (ISA 7.12.2), contiguous-K, unguarded ----
// A 16x32 f16, A row-major with row stride sm (halves), contiguous in k.
// lane L: row m = m0 + L%16 ; kb = (L>=16)?8:0 ; elems = k0+kb+{0..7, 16..23}
__device__ __forceinline__ v16h fragA_g(const _Float16* A, long long sm,
                                        int m0, int k0, int lane){
  const _Float16* p = A + (long long)(m0 + (lane & 15))*sm + k0 + ((lane & 16) ? 8 : 0);
  union { v16h v; v8h h[2]; } u;
  u.h[0] = *(const v8h*)p;
  u.h[1] = *(const v8h*)(p + 16);
  return u.v;
}
// B 32x16 f16 stored N-major: Bt[n][k], row stride sn (halves), contiguous in k.
// lane L: col n = n0 + L%16 ; kb = (L>=16)?8:0 ; elems = k0+kb+{0..7, 16..23}
__device__ __forceinline__ v16h fragB_g(const _Float16* Bt, long long sn,
                                        int k0, int n0, int lane){
  const _Float16* p = Bt + (long long)(n0 + (lane & 15))*sn + k0 + ((lane & 16) ? 8 : 0);
  union { v16h v; v8h h[2]; } u;
  u.h[0] = *(const v8h*)p;
  u.h[1] = *(const v8h*)(p + 16);
  return u.v;
}
// C/D 16x16 f32: VGPR r, lane L -> M = r + ((L>=16)?8:0), N = L%16

// ---- prep kernels ----
__global__ void k_basis(_Float16* bc, _Float16* bs){
  int i = blockIdx.x*blockDim.x + threadIdx.x;
  if (i >= NBP*NFFT) return;
  int j = i / NFFT, kk = i % NFFT;          // [bin][time]  (N-major for fragB)
  float c = 0.f, s = 0.f;
  if (j < NB){
    float wn = 0.5f*(1.f - cosf(6.28318530717958647692f*(float)kk/(float)NFFT));
    float ang = 6.28318530717958647692f * (float)j * (float)kk / (float)NFFT;
    c = wn*cosf(ang); s = wn*sinf(ang);
  }
  bc[i] = (_Float16)c; bs[i] = (_Float16)s;
}

__global__ void k_frames(const float* x, _Float16* fr){
  long long i = (long long)blockIdx.x*blockDim.x + threadIdx.x;
  if (i >= (long long)FB*NFFT) return;
  int n = (int)(i & (NFFT-1));
  long long row = i >> 9;
  int b = (int)(row & (NBAT-1));
  int f = (int)(row >> 3);
  fr[i] = (_Float16)x[(long long)b*TAUD + (long long)f*HOPK + n];
}

__global__ void k_pack_fc1(const float* w, _Float16* dst){   // (512,257) -> (512,288)
  int i = blockIdx.x*blockDim.x + threadIdx.x;
  if (i >= HIDN*NBP) return;
  int n = i / NBP, k = i % NBP;
  dst[i] = (_Float16)(k < NB ? w[n*NB + k] : 0.f);
}

__global__ void k_cvt_h(_Float16* dst, const float* src, long long n){
  long long i = (long long)blockIdx.x*blockDim.x + threadIdx.x;
  if (i < n) dst[i] = (_Float16)src[i];
}

// ---- STFT magnitude: frames(FB,512) x basis^T, fused normalize; 4 waves/block ----
__global__ __launch_bounds__(128) void k_stft(const _Float16* fr, const _Float16* bc,
                       const _Float16* bs, const float* mean, const float* scale,
                       _Float16* magh){
  int lane = threadIdx.x & 31, wave = threadIdx.x >> 5;
  const int tilesN = NBP/16;                 // 18
  int tm = blockIdx.x / tilesN, tn = blockIdx.x % tilesN;
  int m0 = (tm*4 + wave)*16, n0 = tn*16;
  v8f ac = {}; v8f as = {};
#pragma unroll
  for (int k0 = 0; k0 < NFFT; k0 += 32){
    v16h a  = fragA_g(fr, NFFT, m0, k0, lane);
    v16h b1 = fragB_g(bc, NFFT, k0, n0, lane);
    v16h b2 = fragB_g(bs, NFFT, k0, n0, lane);
    ac = __builtin_amdgcn_wmma_f32_16x16x32_f16(false, a, false, b1, (short)0, ac, false, false);
    as = __builtin_amdgcn_wmma_f32_16x16x32_f16(false, a, false, b2, (short)0, as, false, false);
  }
#pragma unroll
  for (int r = 0; r < 8; ++r){
    int m = m0 + r + ((lane & 16) >> 1);
    int n = n0 + (lane & 15);
    float v = 0.f;
    if (n < NB){
      float mg = sqrtf(ac[r]*ac[r] + as[r]*as[r]);
      v = (mg + mean[n]) * scale[n];
    }
    magh[(long long)m*NBP + n] = (_Float16)v;
  }
}

// ---- generic batched WMMA GEMM, K compile-time, 64x32 per block (4 waves) ----
// A: row-major [m][k] row stride asm_, batch stride abs_
// Bt: N-major  [n][k] row stride bsn_, batch stride bbs_
// C: [m][n] contiguous n, row stride csm_, batch stride cbs_ (f32 and/or f16)
template <int K>
__global__ __launch_bounds__(128) void k_gemm(const _Float16* __restrict__ A, long long abs_, long long asm_,
                       const _Float16* __restrict__ Bt, long long bbs_, long long bsn_,
                       float* __restrict__ Cf, _Float16* __restrict__ Ch,
                       long long cbs_, long long csm_, int tilesN2){
  int lane = threadIdx.x & 31, wave = threadIdx.x >> 5;
  int bt = blockIdx.y;
  int tm = blockIdx.x / tilesN2, tn = blockIdx.x % tilesN2;
  int m0 = (tm*4 + wave)*16, n0 = tn*32;
  const _Float16* Ap = A  + (long long)bt*abs_;
  const _Float16* Bp = Bt + (long long)bt*bbs_;
  v8f acc0 = {}, acc1 = {};
#pragma unroll
  for (int k0 = 0; k0 < K; k0 += 32){
    v16h a  = fragA_g(Ap, asm_, m0, k0, lane);
    v16h b0 = fragB_g(Bp, bsn_, k0, n0, lane);
    v16h b1 = fragB_g(Bp, bsn_, k0, n0+16, lane);
    acc0 = __builtin_amdgcn_wmma_f32_16x16x32_f16(false, a, false, b0, (short)0, acc0, false, false);
    acc1 = __builtin_amdgcn_wmma_f32_16x16x32_f16(false, a, false, b1, (short)0, acc1, false, false);
  }
  int mr = m0 + ((lane & 16) >> 1);
  int nc = n0 + (lane & 15);
  if (Cf){
    float* c0 = Cf + (long long)bt*cbs_ + (long long)mr*csm_ + nc;
#pragma unroll
    for (int r = 0; r < 8; ++r){ c0[(long long)r*csm_] = acc0[r]; c0[(long long)r*csm_ + 16] = acc1[r]; }
  }
  if (Ch){
    _Float16* c0 = Ch + (long long)bt*cbs_ + (long long)mr*csm_ + nc;
#pragma unroll
    for (int r = 0; r < 8; ++r){ c0[(long long)r*csm_] = (_Float16)acc0[r]; c0[(long long)r*csm_ + 16] = (_Float16)acc1[r]; }
  }
}

// ---- BatchNorm (batch stats) + tanh ----
__global__ void k_bn_stats(const float* x, float* stats){
  __shared__ float s1[256], s2[256];
  int ch = blockIdx.x, tid = threadIdx.x;
  float a = 0.f, b = 0.f;
  for (int r = tid; r < FB; r += 256){
    float v = x[(long long)r*HIDN + ch]; a += v; b += v*v;
  }
  s1[tid] = a; s2[tid] = b; __syncthreads();
  for (int st = 128; st > 0; st >>= 1){
    if (tid < st){ s1[tid] += s1[tid+st]; s2[tid] += s2[tid+st]; }
    __syncthreads();
  }
  if (tid == 0){
    float mu = s1[0]/(float)FB;
    float var = s2[0]/(float)FB - mu*mu;
    stats[ch] = mu; stats[HIDN+ch] = rsqrtf(var + 1e-5f);
  }
}

__global__ void k_bn_tanh(const float* x, const float* stats, const float* g,
                          const float* bb, _Float16* out){
  long long i = (long long)blockIdx.x*blockDim.x + threadIdx.x;
  if (i >= (long long)FB*HIDN) return;
  int ch = (int)(i & (HIDN-1));
  float v = g[ch]*(x[i]-stats[ch])*stats[HIDN+ch] + bb[ch];
  out[i] = (_Float16)tanhf(v);
}

// ---- persistent text biLSTM: one WG per direction, WMMA h@Whh^T from LDS ----
__global__ __launch_bounds__(1024) void k_text_lstm(const float* wih, const int* idx,
                  const _Float16* whh_h, const float* bias, _Float16* hout){
  __shared__ _Float16 sh_h[16][H2];      // rows 8..15 stay zero (pad to full tile)
  __shared__ float    sh_c[NBAT][H2];
  __shared__ float    sh_g[NBAT][G4];
  int d = blockIdx.x;
  int tid = threadIdx.x, wave = tid >> 5, lane = tid & 31;
  const float*    wihd = wih + (long long)d*G4*NVOC;
  const _Float16* wt   = whh_h + (long long)d*G4*H2;   // N-major (4H, H2) as given
  const float*    bd   = bias + d*G4;
  for (int u = tid; u < 16*H2; u += 1024) sh_h[u>>8][u&255] = (_Float16)0.f;
  for (int u = tid; u < NBAT*H2; u += 1024) sh_c[u>>8][u&255] = 0.f;
  __syncthreads();
  for (int step = 0; step < LTXT; ++step){
    int t = d ? (LTXT-1-step) : step;
    {
      int n0 = wave*32;                      // 32 waves x 32 cols = 1024 gates
      v8f acc0 = {}, acc1 = {};
#pragma unroll
      for (int k0 = 0; k0 < H2; k0 += 32){
        v16h a  = fragA_g(&sh_h[0][0], H2, 0, k0, lane);
        v16h b0 = fragB_g(wt, H2, k0, n0, lane);
        v16h b1 = fragB_g(wt, H2, k0, n0+16, lane);
        acc0 = __builtin_amdgcn_wmma_f32_16x16x32_f16(false, a, false, b0, (short)0, acc0, false, false);
        acc1 = __builtin_amdgcn_wmma_f32_16x16x32_f16(false, a, false, b1, (short)0, acc1, false, false);
      }
#pragma unroll
      for (int r = 0; r < 8; ++r){
        int m = r + ((lane & 16) >> 1);
        if (m < NBAT){
          int n = n0 + (lane & 15);
          // one-hot input projection == column gather of Wih
          sh_g[m][n]    = acc0[r] + bd[n]    + wihd[(long long)n*NVOC      + idx[m*LTXT + t]];
          sh_g[m][n+16] = acc1[r] + bd[n+16] + wihd[(long long)(n+16)*NVOC + idx[m*LTXT + t]];
        }
      }
    }
    __syncthreads();
    for (int u = tid; u < NBAT*H2; u += 1024){
      int b = u >> 8, j = u & 255;
      float gi = sh_g[b][j], gf = sh_g[b][H2+j], gg = sh_g[b][2*H2+j], go = sh_g[b][3*H2+j];
      float c = sigf(gf)*sh_c[b][j] + sigf(gi)*tanhf(gg);
      float h = sigf(go)*tanhf(c);
      sh_c[b][j] = c; sh_h[b][j] = (_Float16)h;
      hout[((long long)t*NBAT + b)*HIDN + d*H2 + j] = (_Float16)h;
    }
    __syncthreads();
  }
}

// ---- persistent audio LSTM: precomputed input projections (f16 pre) ----
__global__ __launch_bounds__(1024) void k_audio_lstm(const _Float16* pre, const _Float16* whh_h,
                  const float* bias, _Float16* xout){
  __shared__ _Float16 sh_h[16][H2];
  __shared__ float    sh_c[NBAT][H2];
  __shared__ float    sh_g[NBAT][G4];
  int d = blockIdx.x;
  int tid = threadIdx.x, wave = tid >> 5, lane = tid & 31;
  const _Float16* pred = pre + (long long)d*FB*G4;
  const _Float16* wt   = whh_h + (long long)d*G4*H2;
  const float*    bd   = bias + d*G4;
  for (int u = tid; u < 16*H2; u += 1024) sh_h[u>>8][u&255] = (_Float16)0.f;
  for (int u = tid; u < NBAT*H2; u += 1024) sh_c[u>>8][u&255] = 0.f;
  __syncthreads();
  for (int step = 0; step < NFR; ++step){
    int t = d ? (NFR-1-step) : step;
    {
      int n0 = wave*32;
      v8f acc0 = {}, acc1 = {};
#pragma unroll
      for (int k0 = 0; k0 < H2; k0 += 32){
        v16h a  = fragA_g(&sh_h[0][0], H2, 0, k0, lane);
        v16h b0 = fragB_g(wt, H2, k0, n0, lane);
        v16h b1 = fragB_g(wt, H2, k0, n0+16, lane);
        acc0 = __builtin_amdgcn_wmma_f32_16x16x32_f16(false, a, false, b0, (short)0, acc0, false, false);
        acc1 = __builtin_amdgcn_wmma_f32_16x16x32_f16(false, a, false, b1, (short)0, acc1, false, false);
      }
#pragma unroll
      for (int r = 0; r < 8; ++r){
        int m = r + ((lane & 16) >> 1);
        if (m < NBAT){
          int n = n0 + (lane & 15);
          long long pr = ((long long)t*NBAT + m)*G4;
          sh_g[m][n]    = acc0[r] + bd[n]    + (float)pred[pr + n];
          sh_g[m][n+16] = acc1[r] + bd[n+16] + (float)pred[pr + n + 16];
        }
      }
    }
    __syncthreads();
    for (int u = tid; u < NBAT*H2; u += 1024){
      int b = u >> 8, j = u & 255;
      float gi = sh_g[b][j], gf = sh_g[b][H2+j], gg = sh_g[b][2*H2+j], go = sh_g[b][3*H2+j];
      float c = sigf(gf)*sh_c[b][j] + sigf(gi)*tanhf(gg);
      float h = sigf(go)*tanhf(c);
      sh_c[b][j] = c; sh_h[b][j] = (_Float16)h;
      xout[((long long)t*NBAT + b)*HIDN + d*H2 + j] = (_Float16)h;
    }
    __syncthreads();
  }
}

// ---- DTW row scan ----
__global__ void k_dtw(const float* scores, float* out){
  __shared__ float prev[LTXT+1];
  int b = blockIdx.x, m = threadIdx.x;
  if (m == 0) prev[0] = POSV;
  prev[m+1] = NEGV;
  __syncthreads();
  const float* sb = scores + (long long)b*NFR*LTXT;
  float* ob = out + (long long)b*NFR*LTXT;
  for (int n = 0; n < NFR; ++n){
    float s = sb[(long long)n*LTXT + m];
    float v = s + fmaxf(prev[m+1], prev[m]);
    __syncthreads();
    prev[m+1] = v;
    if (m == 0) prev[0] = NEGV;
    ob[(long long)n*LTXT + m] = v;
    __syncthreads();
  }
}

extern "C" void kernel_launch(void* const* d_in, const int* in_sizes, int n_in,
                              void* d_out, int out_size, void* d_ws, size_t ws_size,
                              hipStream_t stream){
  const float* x_audio  = (const float*)d_in[0];
  const int*   text_idx = (const int*)  d_in[1];
  const float* txt_wih  = (const float*)d_in[2];
  const float* txt_whh  = (const float*)d_in[3];
  const float* txt_b    = (const float*)d_in[4];
  const float* w_s      = (const float*)d_in[5];
  const float* fc1_w    = (const float*)d_in[6];
  const float* bn1_g    = (const float*)d_in[7];
  const float* bn1_b    = (const float*)d_in[8];
  const float* in_mean  = (const float*)d_in[9];
  const float* in_scale = (const float*)d_in[10];
  const float* ae_wih   = (const float*)d_in[11];
  const float* ae_whh   = (const float*)d_in[12];
  const float* ae_b     = (const float*)d_in[13];
  float* out = (float*)d_out;
  (void)in_sizes; (void)n_in; (void)out_size; (void)ws_size;

  char* w = (char*)d_ws;
  size_t off = 0;
  auto alloc = [&](size_t bytes)->char*{
    char* p = w + off; off += (bytes + 255) & ~(size_t)255; return p;
  };
  // R1: LSTM input-projection buffers (2 dirs); phase-1 reuse for frames/magh
  _Float16* pre    = (_Float16*)alloc((size_t)2*FB*G4*sizeof(_Float16)); // 67 MB
  _Float16* frames = pre;                     // 16.8 MB < 33.6 MB half
  _Float16* magh   = pre + (size_t)FB*G4;     // FB*288*2 = 9.4 MB < 33.6 MB half
  // R2: front-end f32 activations; later reused for scores
  float* xfe    = (float*)alloc((size_t)FB*HIDN*sizeof(float));          // 33.6 MB
  float* scores = xfe;
  // persistent buffers (all N-major f16 weight images)
  _Float16* basisC    = (_Float16*)alloc((size_t)NBP*NFFT*2);
  _Float16* basisS    = (_Float16*)alloc((size_t)NBP*NFFT*2);
  _Float16* fc1h      = (_Float16*)alloc((size_t)HIDN*NBP*2);
  float*    stats     = (float*)   alloc((size_t)2*HIDN*4);
  _Float16* xh0       = (_Float16*)alloc((size_t)FB*HIDN*2);
  _Float16* xh1       = (_Float16*)alloc((size_t)FB*HIDN*2);
  _Float16* xh2       = (_Float16*)alloc((size_t)FB*HIDN*2);
  _Float16* txt_whh_h = (_Float16*)alloc((size_t)2*G4*H2*2);
  _Float16* ae_wih_h  = (_Float16*)alloc((size_t)4*G4*HIDN*2);
  _Float16* ae_whh_h  = (_Float16*)alloc((size_t)4*G4*H2*2);
  _Float16* wsh       = (_Float16*)alloc((size_t)HIDN*HIDN*2);
  _Float16* htext     = (_Float16*)alloc((size_t)LTXT*NBAT*HIDN*2);
  _Float16* sideT     = (_Float16*)alloc((size_t)NBAT*LTXT*HIDN*2);      // [b][l][h]

  // prep: basis + weight converts (layouts already N-major; no transposes)
  k_basis<<<(NBP*NFFT+255)/256, 256, 0, stream>>>(basisC, basisS);
  k_pack_fc1<<<(HIDN*NBP+255)/256, 256, 0, stream>>>(fc1_w, fc1h);
  k_cvt_h<<<(2*G4*H2+255)/256, 256, 0, stream>>>(txt_whh_h, txt_whh, (long long)2*G4*H2);
  k_cvt_h<<<(4*G4*HIDN+255)/256, 256, 0, stream>>>(ae_wih_h, ae_wih, (long long)4*G4*HIDN);
  k_cvt_h<<<(4*G4*H2+255)/256, 256, 0, stream>>>(ae_whh_h, ae_whh, (long long)4*G4*H2);
  k_cvt_h<<<(HIDN*HIDN+255)/256, 256, 0, stream>>>(wsh, w_s, (long long)HIDN*HIDN);

  // STFT magnitude (dual WMMA) + normalize, padded to 288 bins
  k_frames<<<(FB*NFFT+255)/256, 256, 0, stream>>>(x_audio, frames);
  k_stft<<<(FB/64)*(NBP/16), 128, 0, stream>>>(frames, basisC, basisS, in_mean, in_scale, magh);

  // fc1: (FB,288) x (288,512) -> f32
  k_gemm<NBP><<<dim3((FB/64)*(HIDN/32), 1), 128, 0, stream>>>(
      magh, 0, NBP,  fc1h, 0, NBP,
      xfe, nullptr, 0, HIDN, HIDN/32);

  // BN (batch stats) + tanh -> f16
  k_bn_stats<<<HIDN, 256, 0, stream>>>(xfe, stats);
  k_bn_tanh<<<(FB*HIDN+255)/256, 256, 0, stream>>>(xfe, stats, bn1_g, bn1_b, xh0);

  // text biLSTM (persistent, one WG per direction)
  k_text_lstm<<<2, 1024, 0, stream>>>(txt_wih, text_idx, txt_whh_h, txt_b, htext);

  // sideT[b] = htext[b] (512l,512k) x w_s^T -> [b][l][h] f16  (== (w_s @ h^T)^T)
  k_gemm<HIDN><<<dim3((LTXT/64)*(HIDN/32), NBAT), 128, 0, stream>>>(
      htext, HIDN /*b off*/, (long long)NBAT*HIDN /*l stride*/,
      wsh, 0, HIDN,
      nullptr, sideT, (long long)LTXT*HIDN, HIDN, HIDN/32);

  // audio encoder: 2 layers x (batched input-proj GEMMs + persistent recurrence)
  const _Float16* xin = xh0;
  _Float16* xouts[2] = { xh1, xh2 };
  for (int l = 0; l < 2; ++l){
    for (int d = 0; d < 2; ++d){
      k_gemm<HIDN><<<dim3((FB/64)*(G4/32), 1), 128, 0, stream>>>(
          xin, 0, HIDN,
          ae_wih_h + (size_t)(l*2+d)*G4*HIDN, 0, HIDN,
          nullptr, pre + (size_t)d*FB*G4, 0, G4, G4/32);
    }
    k_audio_lstm<<<2, 1024, 0, stream>>>(
        pre, ae_whh_h + (size_t)l*2*G4*H2, ae_b + l*2*G4, xouts[l]);
    xin = xouts[l];
  }

  // scores[b] = x_enc[b] (2048,512) x sideT[b]^T -> f32 (B as [l][h], contiguous k)
  k_gemm<HIDN><<<dim3((NFR/64)*(LTXT/32), NBAT), 128, 0, stream>>>(
      xh2, HIDN /*b off*/, (long long)NBAT*HIDN,
      sideT, (long long)LTXT*HIDN, HIDN,
      scores, nullptr, (long long)NFR*LTXT, LTXT, LTXT/32);

  // DTW scan -> d_out (B, 2048, 512)
  k_dtw<<<NBAT, LTXT, 0, stream>>>(scores, out);
}
